// D_Commune_45329084842140
// MI455X (gfx1250) — matrix operand order repaired and checked by reading, not compiled, not run on hardware
//
#include <hip/hip_runtime.h>
#include <hip/hip_bf16.h>
#include <cstdint>

// ---------------------------------------------------------------------------
// Causal multi-head attention block for MI455X (gfx1250, wave32, WMMA + TDM).
//   B=2, T=2048, C=1024, H=16, D=64
// Pipeline:
//   1) fp32 -> f16 conversion of x, w_attn, w_proj
//   2) QKV GEMM: per-block 16x512 tile; A k-slices DMA'd to LDS by the
//      Tensor Data Mover (double-buffered, s_wait_tensorcnt), f16 WMMA,
//      fused bias / Q-scale / head-split / V-transpose epilogue
//   3) flash-attention: per-wave 16-query tile, 32-key blocks, online softmax
//   4) output projection GEMM (same TDM-fed scheme), fp32 result + bias
// ---------------------------------------------------------------------------

typedef __attribute__((ext_vector_type(16))) _Float16     v16h;
typedef __attribute__((ext_vector_type(8)))  _Float16     v8h;
typedef __attribute__((ext_vector_type(8)))  float        v8f;
typedef __attribute__((ext_vector_type(4)))  unsigned int v4u;
typedef __attribute__((ext_vector_type(8)))  int          v8i;
typedef __attribute__((ext_vector_type(4)))  int          v4i;

__device__ __forceinline__ v8f wmma_f32_16x16x32_f16(v16h a, v16h b, v8f c) {
  return __builtin_amdgcn_wmma_f32_16x16x32_f16(
      /*neg_a=*/false, a, /*neg_b=*/false, b,
      /*c_mod=*/(short)0, c, /*reuse_a=*/false, /*reuse_b=*/false);
}

__device__ __forceinline__ v16h cat16(v8h lo, v8h hi) {
  return __builtin_shufflevector(lo, hi, 0, 1, 2, 3, 4, 5, 6, 7,
                                         8, 9, 10, 11, 12, 13, 14, 15);
}

// A fragment: 16x32 f16, row-major source, leading dim lda (ISA 7.12.2).
__device__ __forceinline__ v16h load_frag_a(const _Float16* src, int lda) {
  const int lane = (int)(threadIdx.x & 31u);
  const _Float16* p = src + (lane & 15) * lda + ((lane >> 4) << 3);
  v8h lo = *(const v8h*)(p);
  v8h hi = *(const v8h*)(p + 16);
  return cat16(lo, hi);
}

// B fragment: (K=32) x (N=16) operand; memory = N rows of contiguous K.
__device__ __forceinline__ v16h load_frag_b(const _Float16* src, int ldb) {
  const int lane = (int)(threadIdx.x & 31u);
  const _Float16* p = src + (lane & 15) * ldb + ((lane >> 4) << 3);
  v8h lo = *(const v8h*)(p);
  v8h hi = *(const v8h*)(p + 16);
  return cat16(lo, hi);
}

// ---------------------------------------------------------------------------
// Tensor Data Mover: DMA a 2D f16 tile (tile_w x tile_h, row stride stride_w
// elements) from global memory into LDS at byte offset lds_off.
// D# built per CDNA5 ISA ch.8: group0 = {count, lds_addr, global_addr, type=2},
// group1 = {data_size=2B, tensor dims, tile dims, dim0 stride}.
// ---------------------------------------------------------------------------
__device__ __forceinline__ void tdm_load_2d_f16(unsigned lds_off,
                                                const _Float16* gsrc,
                                                unsigned tile_w, unsigned tile_h,
                                                unsigned stride_w) {
  const unsigned long long ga = (unsigned long long)(uintptr_t)gsrc;
  v4u g0;
  g0.x = 1u;                                   // count=1, user mode, no gather
  g0.y = lds_off;                              // lds_addr (bytes)
  g0.z = (unsigned)(ga & 0xffffffffu);         // global_addr[31:0]
  g0.w = (unsigned)((ga >> 32) & 0x01ffffffu)  // global_addr[56:32]
         | (2u << 30);                         // type = 2 ("image")
  v8i g1;
  g1[0] = (int)(1u << 16);                     // wg_mask=0, data_size=1 (2B)
  g1[1] = (int)((tile_w & 0xffffu) << 16);     // tensor_dim0[15:0] = tile_w
  g1[2] = (int)(((tile_w >> 16) & 0xffffu) |
                ((tile_h & 0xffffu) << 16));   // dim0 hi | tensor_dim1 lo
  g1[3] = (int)(((tile_h >> 16) & 0xffffu) |
                ((tile_w & 0xffffu) << 16));   // dim1 hi | tile_dim0
  g1[4] = (int)(tile_h & 0xffffu);             // tile_dim1 (tile_dim2 = 0)
  g1[5] = (int)stride_w;                       // tensor_dim0_stride[31:0]
  g1[6] = 0;                                   // stride hi, dim1_stride lo
  g1[7] = 0;
  const v4i gz = {0, 0, 0, 0};
#if __clang_major__ >= 23
  const v8i gz8 = {0, 0, 0, 0, 0, 0, 0, 0};
  __builtin_amdgcn_tensor_load_to_lds(g0, g1, gz, gz, gz8, 0);
#else
  __builtin_amdgcn_tensor_load_to_lds(g0, g1, gz, gz, 0);
#endif
}

// ---------------------------------------------------------------------------
// 1) fp32 -> f16 conversion, 8 elements / thread (b128 in, b128 out)
// ---------------------------------------------------------------------------
__global__ void __launch_bounds__(256)
cvt_f32_to_f16_kernel(const float* __restrict__ in, _Float16* __restrict__ out,
                      int n) {
  int i = (int)(blockIdx.x * blockDim.x + threadIdx.x) * 8;
  if (i + 8 > n) return;
  float4 a = *(const float4*)(in + i);
  float4 b = *(const float4*)(in + i + 4);
  v8h o = {(_Float16)a.x, (_Float16)a.y, (_Float16)a.z, (_Float16)a.w,
           (_Float16)b.x, (_Float16)b.y, (_Float16)b.z, (_Float16)b.w};
  *(v8h*)(out + i) = o;
}

// ---------------------------------------------------------------------------
// 2) QKV GEMM: kqv[m,n] = sum_k xh[m,k] * wh[n,k] + bias[n]
//    M=4096, N=3072, K=1024. Block = 8 waves = 16 rows x 512 cols; the shared
//    16x32 A k-slice is TDM-DMA'd into double-buffered LDS once per block.
//    Epilogue: K -> (B,H,T,D), Q -> (B,H,T,D) * D^-0.5, V -> (B,H,D,T).
// ---------------------------------------------------------------------------
__global__ void __launch_bounds__(256)
qkv_gemm_kernel(const _Float16* __restrict__ xh,   // 4096 x 1024
                const _Float16* __restrict__ wh,   // 3072 x 1024
                const float* __restrict__ bias,    // 3072
                _Float16* __restrict__ Kh,         // (32, 2048, 64)
                _Float16* __restrict__ Qh,         // (32, 2048, 64)
                _Float16* __restrict__ Vt) {       // (32, 64, 2048)
  __shared__ _Float16 atile[2][16 * 32];           // double-buffered A slice

  const int tid = (int)threadIdx.x;
  const int wid = tid >> 5;   // 0..7
  const int lane = tid & 31;
  const int NB = 3072 / 512;  // 6 column blocks
  const int mt = (int)blockIdx.x / NB;
  const int nb = (int)blockIdx.x % NB;
  const int m0 = mt * 16;
  const int n0 = nb * 512 + wid * 64;

  const v8f zero = {0.f, 0.f, 0.f, 0.f, 0.f, 0.f, 0.f, 0.f};
  v8f acc[4] = {zero, zero, zero, zero};

  const _Float16* arow = xh + (size_t)m0 * 1024;
  if (wid == 0)
    tdm_load_2d_f16((unsigned)(uintptr_t)&atile[0][0], arow, 32, 16, 1024);

  for (int i = 0; i < 32; ++i) {  // 1024 / 32 k-slices
    const int k0 = i * 32;
    if (wid == 0) {
      if (i + 1 < 32) {
        tdm_load_2d_f16((unsigned)(uintptr_t)&atile[(i + 1) & 1][0],
                        arow + (i + 1) * 32, 32, 16, 1024);
        __builtin_amdgcn_s_wait_tensorcnt((short)1);  // slice i landed
      } else {
        __builtin_amdgcn_s_wait_tensorcnt((short)0);
      }
    }
    __syncthreads();  // A slice visible to all 8 waves
    v16h a = load_frag_a(&atile[i & 1][0], 32);  // ds_load_b128
#pragma unroll
    for (int j = 0; j < 4; ++j) {
      v16h b = load_frag_b(wh + (size_t)(n0 + j * 16) * 1024 + k0, 1024);
      acc[j] = wmma_f32_16x16x32_f16(a, b, acc[j]);
    }
    __syncthreads();  // reads done before buffer is overwritten
  }

  // C/D layout: row m = m0 + e + 8*(lane>=16), col n = n0 + j*16 + (lane&15)
  const int hi8 = (lane >> 4) << 3;
  const int col = lane & 15;
#pragma unroll
  for (int j = 0; j < 4; ++j) {
    const int n = n0 + j * 16 + col;
    const int which = n >> 10;  // 0=K, 1=Q, 2=V (strip never crosses boundary)
    const int c = n & 1023;
    const int h = c >> 6;
    const int d = c & 63;
    const float bv = bias[n];
#pragma unroll
    for (int e = 0; e < 8; ++e) {
      const int m = m0 + hi8 + e;
      const int b = m >> 11;
      const int t = m & 2047;
      const size_t bh = (size_t)(b * 16 + h);
      const float v = acc[j][e] + bv;
      if (which == 0)
        Kh[(bh * 2048 + t) * 64 + d] = (_Float16)v;
      else if (which == 1)
        Qh[(bh * 2048 + t) * 64 + d] = (_Float16)(v * 0.125f);  // 1/sqrt(64)
      else
        Vt[(bh * 64 + d) * 2048 + t] = (_Float16)v;  // transposed for PV GEMM
    }
  }
}

// ---------------------------------------------------------------------------
// 3) Flash attention: one wave per (b,h, 16-query tile); 32 keys per step.
//    S = Q K^T (Q pre-scaled), causal mask, online softmax, O += P V.
// ---------------------------------------------------------------------------
__global__ void __launch_bounds__(128)
attn_kernel(const _Float16* __restrict__ Qh, const _Float16* __restrict__ Kh,
            const _Float16* __restrict__ Vt, _Float16* __restrict__ ah) {
  __shared__ _Float16 pbuf[4][16 * 32];  // per-wave P transpose staging (1KB)

  const int lane = (int)(threadIdx.x & 31u);
  const int wslot = (int)(threadIdx.x >> 5);
  const int wave = (int)blockIdx.x * 4 + wslot;
  const int qt = wave & 127;  // query tile (T/16 = 128)
  const int bh = wave >> 7;   // 0..31 = b*16+h

  const _Float16* Qb = Qh + ((size_t)bh * 2048 + (size_t)qt * 16) * 64;
  const _Float16* Kb = Kh + (size_t)bh * 2048 * 64;
  const _Float16* Vb = Vt + (size_t)bh * 64 * 2048;
  _Float16* prow = &pbuf[wslot][0];

  // Q tile 16x64 as two A fragments (d = 0..31 and 32..63), kept in VGPRs
  v16h qa0 = load_frag_a(Qb, 64);
  v16h qa1 = load_frag_a(Qb + 32, 64);

  const int hi8 = (lane >> 4) << 3;
  const int col = lane & 15;

  const v8f zero = {0.f, 0.f, 0.f, 0.f, 0.f, 0.f, 0.f, 0.f};
  v8f o[4] = {zero, zero, zero, zero};  // O accum: 16x64 fp32
  float mrow[8], lrow[8];
#pragma unroll
  for (int e = 0; e < 8; ++e) { mrow[e] = -1e30f; lrow[e] = 0.f; }

  const int nkb = (qt + 2) >> 1;  // ceil((qt+1)*16 / 32) key blocks
  for (int kb = 0; kb < nkb; ++kb) {
    const int kbase = kb << 5;

    // ---- scores: two 16x16 tiles, K-dim = d (2 chunks of 32) ----
    v8f s[2] = {zero, zero};
#pragma unroll
    for (int hf = 0; hf < 2; ++hf) {
      const _Float16* kp = Kb + (size_t)(kbase + hf * 16) * 64;
      s[hf] = wmma_f32_16x16x32_f16(qa0, load_frag_b(kp, 64), s[hf]);
      s[hf] = wmma_f32_16x16x32_f16(qa1, load_frag_b(kp + 32, 64), s[hf]);
    }

    // ---- causal mask + online softmax (row reduce over 16-lane groups) ----
    float scl[8];
#pragma unroll
    for (int e = 0; e < 8; ++e) {
      const int qi = (qt << 4) + hi8 + e;
      float s0 = (kbase + col <= qi) ? s[0][e] : -1e30f;
      float s1 = (kbase + 16 + col <= qi) ? s[1][e] : -1e30f;
      float rm = fmaxf(s0, s1);
      rm = fmaxf(rm, __shfl_xor(rm, 1, 32));
      rm = fmaxf(rm, __shfl_xor(rm, 2, 32));
      rm = fmaxf(rm, __shfl_xor(rm, 4, 32));
      rm = fmaxf(rm, __shfl_xor(rm, 8, 32));
      const float mnew = fmaxf(mrow[e], rm);
      const float p0 = __expf(s0 - mnew);
      const float p1 = __expf(s1 - mnew);
      float rs = p0 + p1;
      rs += __shfl_xor(rs, 1, 32);
      rs += __shfl_xor(rs, 2, 32);
      rs += __shfl_xor(rs, 4, 32);
      rs += __shfl_xor(rs, 8, 32);
      const float f = __expf(mrow[e] - mnew);
      lrow[e] = lrow[e] * f + rs;
      mrow[e] = mnew;
      scl[e] = f;
      // C-layout -> row-major P tile in LDS (f16) for A-fragment reload
      prow[(hi8 + e) * 32 + col] = (_Float16)p0;
      prow[(hi8 + e) * 32 + col + 16] = (_Float16)p1;
    }

    // ---- O = O*scale + P (16x32) x V (32x64) ----
    v16h pa = load_frag_a(prow, 32);  // ds_load_b128, same-wave RAW via DScnt
#pragma unroll
    for (int j = 0; j < 4; ++j) {
#pragma unroll
      for (int e = 0; e < 8; ++e) o[j][e] *= scl[e];
      v16h bv = load_frag_b(Vb + (size_t)(j * 16) * 2048 + kbase, 2048);
      o[j] = wmma_f32_16x16x32_f16(pa, bv, o[j]);
    }
  }

  // ---- normalize + store attn in (B, T, C) f16 for the projection GEMM ----
  const int b = bh >> 4;
  const int h = bh & 15;
#pragma unroll
  for (int j = 0; j < 4; ++j) {
#pragma unroll
    for (int e = 0; e < 8; ++e) {
      const int t = (qt << 4) + hi8 + e;
      const int c = h * 64 + j * 16 + col;
      ah[((size_t)b * 2048 + t) * 1024 + c] = (_Float16)(o[j][e] / lrow[e]);
    }
  }
}

// ---------------------------------------------------------------------------
// 4) Output projection: out[m,n] = sum_k ah[m,k] * wph[n,k] + bias[n] (fp32)
//    M=4096, N=1024, K=1024. Same TDM-fed block structure (16 x 512).
// ---------------------------------------------------------------------------
__global__ void __launch_bounds__(256)
proj_gemm_kernel(const _Float16* __restrict__ ah,   // 4096 x 1024
                 const _Float16* __restrict__ wph,  // 1024 x 1024
                 const float* __restrict__ bias,    // 1024
                 float* __restrict__ out) {         // 4096 x 1024
  __shared__ _Float16 atile[2][16 * 32];

  const int tid = (int)threadIdx.x;
  const int wid = tid >> 5;
  const int lane = tid & 31;
  const int NB = 1024 / 512;  // 2 column blocks
  const int mt = (int)blockIdx.x / NB;
  const int nb = (int)blockIdx.x % NB;
  const int m0 = mt * 16;
  const int n0 = nb * 512 + wid * 64;

  const v8f zero = {0.f, 0.f, 0.f, 0.f, 0.f, 0.f, 0.f, 0.f};
  v8f acc[4] = {zero, zero, zero, zero};

  const _Float16* arow = ah + (size_t)m0 * 1024;
  if (wid == 0)
    tdm_load_2d_f16((unsigned)(uintptr_t)&atile[0][0], arow, 32, 16, 1024);

  for (int i = 0; i < 32; ++i) {
    const int k0 = i * 32;
    if (wid == 0) {
      if (i + 1 < 32) {
        tdm_load_2d_f16((unsigned)(uintptr_t)&atile[(i + 1) & 1][0],
                        arow + (i + 1) * 32, 32, 16, 1024);
        __builtin_amdgcn_s_wait_tensorcnt((short)1);
      } else {
        __builtin_amdgcn_s_wait_tensorcnt((short)0);
      }
    }
    __syncthreads();
    v16h a = load_frag_a(&atile[i & 1][0], 32);
#pragma unroll
    for (int j = 0; j < 4; ++j) {
      v16h b = load_frag_b(wph + (size_t)(n0 + j * 16) * 1024 + k0, 1024);
      acc[j] = wmma_f32_16x16x32_f16(a, b, acc[j]);
    }
    __syncthreads();
  }

  const int hi8 = (lane >> 4) << 3;
  const int col = lane & 15;
#pragma unroll
  for (int j = 0; j < 4; ++j) {
    const int n = n0 + j * 16 + col;
    const float bv = bias[n];
#pragma unroll
    for (int e = 0; e < 8; ++e) {
      const int m = m0 + hi8 + e;
      out[(size_t)m * 1024 + n] = acc[j][e] + bv;
    }
  }
}

// ---------------------------------------------------------------------------
extern "C" void kernel_launch(void* const* d_in, const int* in_sizes, int n_in,
                              void* d_out, int out_size, void* d_ws,
                              size_t ws_size, hipStream_t stream) {
  const float* x      = (const float*)d_in[0];  // (2, 2048, 1024)
  const float* w_attn = (const float*)d_in[1];  // (3072, 1024)
  const float* b_attn = (const float*)d_in[2];  // (3072,)
  const float* w_proj = (const float*)d_in[3];  // (1024, 1024)
  const float* b_proj = (const float*)d_in[4];  // (1024,)
  float* out = (float*)d_out;                   // (2, 2048, 1024)

  char* ws = (char*)d_ws;
  _Float16* xh  = (_Float16*)ws; ws += (size_t)4096 * 1024 * 2;      // 8 MB
  _Float16* wah = (_Float16*)ws; ws += (size_t)3072 * 1024 * 2;      // 6 MB
  _Float16* wph = (_Float16*)ws; ws += (size_t)1024 * 1024 * 2;      // 2 MB
  _Float16* Kh  = (_Float16*)ws; ws += (size_t)32 * 2048 * 64 * 2;   // 8 MB
  _Float16* Qh  = (_Float16*)ws; ws += (size_t)32 * 2048 * 64 * 2;   // 8 MB
  _Float16* Vt  = (_Float16*)ws; ws += (size_t)32 * 2048 * 64 * 2;   // 8 MB
  _Float16* ah  = (_Float16*)ws; ws += (size_t)4096 * 1024 * 2;      // 8 MB

  cvt_f32_to_f16_kernel<<<(4096 * 1024 / 8) / 256, 256, 0, stream>>>(
      x, xh, 4096 * 1024);
  cvt_f32_to_f16_kernel<<<(3072 * 1024 / 8) / 256, 256, 0, stream>>>(
      w_attn, wah, 3072 * 1024);
  cvt_f32_to_f16_kernel<<<(1024 * 1024 / 8) / 256, 256, 0, stream>>>(
      w_proj, wph, 1024 * 1024);

  // 256 M-tiles * 6 column blocks, 256 threads (8 waves) per block
  qkv_gemm_kernel<<<256 * 6, 256, 0, stream>>>(xh, wah, b_attn, Kh, Qh, Vt);
  // 32 (b,h) * 128 query tiles waves, 4 waves per block
  attn_kernel<<<(32 * 128) / 4, 128, 0, stream>>>(Qh, Kh, Vt, ah);
  // 256 M-tiles * 2 column blocks
  proj_gemm_kernel<<<256 * 2, 256, 0, stream>>>(ah, wph, b_proj, out);
}
